// GCNAnomalyDetector_39384850104825
// MI455X (gfx1250) — matrix operand order repaired and checked by reading, not compiled, not run on hardware
//
#include <hip/hip_runtime.h>

// ---------------------------------------------------------------------------
// GCN anomaly detector on gfx1250 (MI455X).
// Dense GEMMs via v_wmma_f32_16x16x32_f16 with pre-packed f16 weight
// fragments (lane-contiguous, loaded as 2x b128); each wave computes all
// column tiles so A is read from HBM exactly once. Transformed node features
// (hw1/hw2) are stored in f16 to halve the edge-gather traffic; aggregation
// accumulates in fp32 via global atomics into L2-resident buffers.
// ---------------------------------------------------------------------------

typedef __attribute__((ext_vector_type(16))) _Float16 v16h;
typedef __attribute__((ext_vector_type(8)))  _Float16 v8h;
typedef __attribute__((ext_vector_type(2)))  _Float16 v2h;
typedef __attribute__((ext_vector_type(8)))  float    v8f;
typedef __attribute__((ext_vector_type(4)))  float    v4f;

static __device__ __forceinline__ v8f wmma32(v16h a, v16h b, v8f c) {
  // (neg_a, A, neg_b, B, c_mod, C, reuse_a, reuse_b)
  return __builtin_amdgcn_wmma_f32_16x16x32_f16(false, a, false, b, (short)0, c,
                                                false, false);
}

// A fragment (16x32) from row-major fp32, row stride ld, tile at (rb, kb).
// Lane layout: half=0 -> K = kb+[0..7] and kb+[16..23]; half=1 -> +8.
static __device__ __forceinline__ v16h load_a(const float* A, int ld, int rb,
                                              int kb, int m, int half) {
  const float* p = A + (size_t)(rb + m) * ld + kb + half * 8;
  v4f f0 = *(const v4f*)(p);
  v4f f1 = *(const v4f*)(p + 4);
  v4f f2 = *(const v4f*)(p + 16);
  v4f f3 = *(const v4f*)(p + 20);
  v16h a;
#pragma unroll
  for (int i = 0; i < 4; ++i) {
    a[i]      = (_Float16)f0[i];
    a[4 + i]  = (_Float16)f1[i];
    a[8 + i]  = (_Float16)f2[i];
    a[12 + i] = (_Float16)f3[i];
  }
  return a;
}

// Same but from row-major f16 (global or LDS).
static __device__ __forceinline__ v16h load_a(const _Float16* A, int ld, int rb,
                                              int kb, int m, int half) {
  const _Float16* p = A + (size_t)(rb + m) * ld + kb + half * 8;
  v8h lo = *(const v8h*)(p);
  v8h hi = *(const v8h*)(p + 16);
  v16h a;
#pragma unroll
  for (int i = 0; i < 8; ++i) { a[i] = lo[i]; a[8 + i] = hi[i]; }
  return a;
}

// Packed B fragment: 32 bytes contiguous per lane -> two 16B vector loads.
static __device__ __forceinline__ v16h load_b_packed(const _Float16* P,
                                                     int tile, int lane) {
  const _Float16* p = P + ((size_t)tile * 32 + lane) * 16;
  v8h lo = *(const v8h*)(p);
  v8h hi = *(const v8h*)(p + 8);
  v16h b;
#pragma unroll
  for (int i = 0; i < 8; ++i) { b[i] = lo[i]; b[8 + i] = hi[i]; }
  return b;
}

// One-time weight pack: W[K x Dout] fp32 row-major -> fragment-major f16.
// Tile t = kt*(Dout/16)+nt holds the 32x16 slice at (kt*32, nt*16); lane L
// stores its 16 B-fragment elements contiguously (ISA B layout: column
// nb+(L&15), rows kb+(L>>4)*8+[0..7] and kb+16+(L>>4)*8+[0..7]).
__global__ void pack_b(const float* __restrict__ W, _Float16* __restrict__ out,
                       int K, int Dout) {
  const int tid = blockIdx.x * blockDim.x + threadIdx.x;
  const int tile = tid >> 5;
  const int ntiles = (K / 32) * (Dout / 16);
  if (tile >= ntiles) return;
  const int lane = tid & 31;
  const int m = lane & 15;
  const int half = lane >> 4;
  const int nct = Dout / 16;
  const int kt = tile / nct, nt = tile % nct;
  const float* p = W + (size_t)(kt * 32 + half * 8) * Dout + nt * 16 + m;
  _Float16* o = out + ((size_t)tile * 32 + lane) * 16;
#pragma unroll
  for (int j = 0; j < 16; ++j) o[j] = (_Float16)p[(j + (j >> 3) * 8) * Dout];
}

// ---------------------------------------------------------------------------
// WMMA GEMM: C[nrows x Dout] = A[nrows x K] * W (packed); f16 output.
// One wave per 16-row tile computes ALL NT column tiles, so each A fragment
// is fetched from HBM exactly once. nrows % 16 == 0, Dout == NT*16.
// ---------------------------------------------------------------------------
template <int NT, typename T>
__global__ void wmma_gemm(const T* __restrict__ A,
                          const _Float16* __restrict__ Bp,
                          _Float16* __restrict__ C, int nrows, int K) {
  const int lane = threadIdx.x & 31;
  const int m = lane & 15;
  const int half = lane >> 4;
  const int wave = threadIdx.x >> 5;
  const int rb = (blockIdx.x * (blockDim.x >> 5) + wave) * 16;
  if (rb >= nrows) return;
  const int Dout = NT * 16;

  v8f acc[NT] = {};
  const int ktiles = K / 32;
  for (int kt = 0; kt < ktiles; ++kt) {
    v16h a = load_a(A, K, rb, kt * 32, m, half);
#pragma unroll
    for (int nt = 0; nt < NT; ++nt) {
      v16h b = load_b_packed(Bp, kt * NT + nt, lane);
      acc[nt] = wmma32(a, b, acc[nt]);
    }
  }
#pragma unroll
  for (int nt = 0; nt < NT; ++nt)
#pragma unroll
    for (int r = 0; r < 8; ++r)
      C[(size_t)(rb + half * 8 + r) * Dout + nt * 16 + m] =
          (_Float16)acc[nt][r];
}

// ---------------------------------------------------------------------------
// Sparse phase
// ---------------------------------------------------------------------------
__global__ void fill_f32(float* __restrict__ p, long long n, float v) {
  long long i = (long long)blockIdx.x * blockDim.x + threadIdx.x;
  if (i < n) p[i] = v;
}

__global__ void degree_k(const int* __restrict__ dst, float* __restrict__ deg,
                         long long E) {
  long long e = (long long)blockIdx.x * blockDim.x + threadIdx.x;
  if (e < E) atomicAdd(&deg[dst[e]], 1.0f);
}

__global__ void rsqrt_k(const float* __restrict__ deg, float* __restrict__ dis,
                        long long n) {
  long long i = (long long)blockIdx.x * blockDim.x + threadIdx.x;
  if (i < n) dis[i] = rsqrtf(deg[i]);
}

// One wave per edge, 2 features per lane (D = 64): coalesced 128B f16 gather,
// fp32 atomic scatter (agg is L2-resident).
__global__ void aggregate64(const _Float16* __restrict__ hw,
                            const int* __restrict__ src,
                            const int* __restrict__ dst,
                            const float* __restrict__ dis,
                            float* __restrict__ agg, long long E) {
  long long tid = (long long)blockIdx.x * blockDim.x + threadIdx.x;
  long long e = tid >> 5;
  if (e >= E) return;
  const int f = (int)(tid & 31) * 2;
  const int s = src[e], d = dst[e];
  const float norm = dis[s] * dis[d];
  const v2h v = *(const v2h*)(hw + (size_t)s * 64 + f);
  atomicAdd(&agg[(size_t)d * 64 + f],     (float)v[0] * norm);
  atomicAdd(&agg[(size_t)d * 64 + f + 1], (float)v[1] * norm);
}

// One wave per edge, 1 feature per lane (D = 32): 64B f16 gather per edge.
__global__ void aggregate32(const _Float16* __restrict__ hw,
                            const int* __restrict__ src,
                            const int* __restrict__ dst,
                            const float* __restrict__ dis,
                            float* __restrict__ agg, long long E) {
  long long tid = (long long)blockIdx.x * blockDim.x + threadIdx.x;
  long long e = tid >> 5;
  if (e >= E) return;
  const int f = (int)(tid & 31);
  const int s = src[e], d = dst[e];
  const float norm = dis[s] * dis[d];
  atomicAdd(&agg[(size_t)d * 32 + f],
            (float)hw[(size_t)s * 32 + f] * norm);
}

// h1 = relu(agg + hw*dis^2 + b); store f16 for next GEMM.
__global__ void finalize_relu16(const float* __restrict__ agg,
                                const _Float16* __restrict__ hw,
                                const float* __restrict__ dis,
                                const float* __restrict__ bias,
                                _Float16* __restrict__ out16, long long total,
                                int D) {
  long long i = (long long)blockIdx.x * blockDim.x + threadIdx.x;
  if (i >= total) return;
  const int n = (int)(i / D), f = (int)(i % D);
  const float d2 = dis[n] * dis[n];
  const float v = agg[i] + (float)hw[i] * d2 + bias[f];
  out16[i] = (_Float16)fmaxf(v, 0.0f);
}

// h = agg + hw*dis^2 + b; write fp32 output + f16 copy for the MLP heads.
__global__ void finalize_out(const float* __restrict__ agg,
                             const _Float16* __restrict__ hw,
                             const float* __restrict__ dis,
                             const float* __restrict__ bias,
                             float* __restrict__ outf,
                             _Float16* __restrict__ out16, long long total,
                             int D) {
  long long i = (long long)blockIdx.x * blockDim.x + threadIdx.x;
  if (i >= total) return;
  const int n = (int)(i / D), f = (int)(i % D);
  const float d2 = dis[n] * dis[n];
  const float v = agg[i] + (float)hw[i] * d2 + bias[f];
  outf[i] = v;
  out16[i] = (_Float16)v;
}

// ---------------------------------------------------------------------------
// Fused MLP heads. One wave per 16-row tile of h. WMMA -> ReLU -> LDS(f16)
// -> WMMA chain, all with pre-packed f16 weight fragments. Per-wave private
// LDS slabs; LDS ops are in-order within a wave, so no barriers needed.
// ---------------------------------------------------------------------------
__global__ void mlp_fused(const _Float16* __restrict__ h16,
                          const _Float16* __restrict__ prW1,  // 4 tiles
                          const float* __restrict__ rb1,
                          const _Float16* __restrict__ prW2,  // 16 tiles
                          const float* __restrict__ rb2,
                          const _Float16* __restrict__ psW1,  // 2 tiles
                          const float* __restrict__ sb1,
                          const float* __restrict__ sW2,
                          const float* __restrict__ sb2,
                          float* __restrict__ xrec, float* __restrict__ score,
                          int nrows) {
  __shared__ __align__(16) _Float16 lds_t1[8][16 * 64];  // relu(h@rW1+rb1)
  __shared__ __align__(16) _Float16 lds_s1[8][16 * 32];  // relu(h@sW1+sb1)

  const int lane = threadIdx.x & 31;
  const int m = lane & 15;
  const int half = lane >> 4;
  const int wave = threadIdx.x >> 5;
  const int rb = (blockIdx.x * (blockDim.x >> 5) + wave) * 16;
  if (rb >= nrows) return;

  _Float16* t1 = lds_t1[wave];
  _Float16* s1 = lds_s1[wave];

  // A fragment of the 16x32 h tile (K = 32, one fragment).
  v16h ah = load_a(h16, 32, rb, 0, m, half);

  // t1 = relu(h @ rW1 + rb1) : [16 x 64]
#pragma unroll
  for (int j = 0; j < 4; ++j) {
    v8f c = {};
    c = wmma32(ah, load_b_packed(prW1, j, lane), c);
    const float bias = rb1[j * 16 + m];
#pragma unroll
    for (int r = 0; r < 8; ++r)
      t1[(r + half * 8) * 64 + j * 16 + m] = (_Float16)fmaxf(c[r] + bias, 0.0f);
  }

  // x_rec = t1 @ rW2 + rb2 : [16 x 128]; rW2 tiles: kt*8 + j.
  v16h a0 = load_a((const _Float16*)t1, 64, 0, 0, m, half);
  v16h a1 = load_a((const _Float16*)t1, 64, 0, 32, m, half);
#pragma unroll
  for (int j = 0; j < 8; ++j) {
    v8f c = {};
    c = wmma32(a0, load_b_packed(prW2, j, lane), c);
    c = wmma32(a1, load_b_packed(prW2, 8 + j, lane), c);
    const float bias = rb2[j * 16 + m];
#pragma unroll
    for (int r = 0; r < 8; ++r)
      xrec[(size_t)(rb + half * 8 + r) * 128 + j * 16 + m] = c[r] + bias;
  }

  // s1 = relu(h @ sW1 + sb1) : [16 x 32]
#pragma unroll
  for (int j = 0; j < 2; ++j) {
    v8f c = {};
    c = wmma32(ah, load_b_packed(psW1, j, lane), c);
    const float bias = sb1[j * 16 + m];
#pragma unroll
    for (int r = 0; r < 8; ++r)
      s1[(r + half * 8) * 32 + j * 16 + m] = (_Float16)fmaxf(c[r] + bias, 0.0f);
  }

  // score = sigmoid(s1 @ sW2 + sb2) : [16 x 1]; lanes 0..15 each do one row.
  if (half == 0) {
    float acc = sb2[0];
#pragma unroll
    for (int k = 0; k < 32; ++k) acc += (float)s1[m * 32 + k] * sW2[k];
    score[rb + m] = 1.0f / (1.0f + expf(-acc));
  }
}

// ---------------------------------------------------------------------------
// Host launch
// ---------------------------------------------------------------------------
extern "C" void kernel_launch(void* const* d_in, const int* in_sizes, int n_in,
                              void* d_out, int out_size, void* d_ws,
                              size_t ws_size, hipStream_t stream) {
  const float* x   = (const float*)d_in[0];
  const int*   ei  = (const int*)d_in[1];
  const float* W1  = (const float*)d_in[2];
  const float* b1  = (const float*)d_in[3];
  const float* W2  = (const float*)d_in[4];
  const float* b2  = (const float*)d_in[5];
  const float* rW1 = (const float*)d_in[6];
  const float* rb1 = (const float*)d_in[7];
  const float* rW2 = (const float*)d_in[8];
  const float* rb2 = (const float*)d_in[9];
  const float* sW1 = (const float*)d_in[10];
  const float* sb1 = (const float*)d_in[11];
  const float* sW2 = (const float*)d_in[12];
  const float* sb2 = (const float*)d_in[13];

  const long long N = in_sizes[0] / 128;   // 100000
  const long long E = in_sizes[1] / 2;     // 3200000
  const int* src = ei;
  const int* dst = ei + E;

  // Workspace carve-up (~78 MB + 43 KB packed weights).
  char* w = (char*)d_ws;
  float*     deg  = (float*)w;     w += N * 4;
  float*     dis  = (float*)w;     w += N * 4;
  _Float16*  hw1  = (_Float16*)w;  w += N * 64 * 2;
  float*     agg1 = (float*)w;     w += N * 64 * 4;
  _Float16*  h1h  = (_Float16*)w;  w += N * 64 * 2;
  _Float16*  hw2  = (_Float16*)w;  w += N * 32 * 2;
  float*     agg2 = (float*)w;     w += N * 32 * 4;
  _Float16*  h16  = (_Float16*)w;  w += N * 32 * 2;
  _Float16*  pW1  = (_Float16*)w;  w += 128 * 64 * 2;  // 16 tiles
  _Float16*  pW2  = (_Float16*)w;  w += 64 * 32 * 2;   //  4 tiles
  _Float16*  prW1 = (_Float16*)w;  w += 32 * 64 * 2;   //  4 tiles
  _Float16*  prW2 = (_Float16*)w;  w += 64 * 128 * 2;  // 16 tiles
  _Float16*  psW1 = (_Float16*)w;  w += 32 * 32 * 2;   //  2 tiles

  // Output partitions: h [N*32] | x_rec [N*128] | score [N].
  float* out_h    = (float*)d_out;
  float* out_xrec = out_h + N * 32;
  float* out_scr  = out_xrec + N * 128;

  const int TB = 256;
  const int row_tiles = (int)(N / 16);                 // N % 16 == 0
  const int gemm_bx = (row_tiles + 7) / 8;             // 8 waves / block
  const int edge_blocks = (int)((E * 32 + TB - 1) / TB);

  hipMemsetAsync(agg1, 0, (size_t)(N * 64 * 4), stream);
  hipMemsetAsync(agg2, 0, (size_t)(N * 32 * 4), stream);

  // One-time weight packing (tiny).
  pack_b<<<1, 512, 0, stream>>>(W1, pW1, 128, 64);
  pack_b<<<1, 128, 0, stream>>>(W2, pW2, 64, 32);
  pack_b<<<1, 128, 0, stream>>>(rW1, prW1, 32, 64);
  pack_b<<<1, 512, 0, stream>>>(rW2, prW2, 64, 128);
  pack_b<<<1, 64, 0, stream>>>(sW1, psW1, 32, 32);

  // Degree (with self-loop) and symmetric norm.
  fill_f32<<<(int)((N + TB - 1) / TB), TB, 0, stream>>>(deg, N, 1.0f);
  degree_k<<<(int)((E + TB - 1) / TB), TB, 0, stream>>>(dst, deg, E);
  rsqrt_k<<<(int)((N + TB - 1) / TB), TB, 0, stream>>>(deg, dis, N);

  // Layer 1: hw1 = x @ W1 ; aggregate ; h1 = relu(agg + hw1*dis^2 + b1).
  wmma_gemm<4, float><<<gemm_bx, TB, 0, stream>>>(x, pW1, hw1, (int)N, 128);
  aggregate64<<<edge_blocks, TB, 0, stream>>>(hw1, src, dst, dis, agg1, E);
  finalize_relu16<<<(int)((N * 64 + TB - 1) / TB), TB, 0, stream>>>(
      agg1, hw1, dis, b1, h1h, N * 64, 64);

  // Layer 2: hw2 = h1 @ W2 ; aggregate ; h = agg + hw2*dis^2 + b2.
  wmma_gemm<2, _Float16><<<gemm_bx, TB, 0, stream>>>(h1h, pW2, hw2, (int)N, 64);
  aggregate32<<<edge_blocks, TB, 0, stream>>>(hw2, src, dst, dis, agg2, E);
  finalize_out<<<(int)((N * 32 + TB - 1) / TB), TB, 0, stream>>>(
      agg2, hw2, dis, b2, out_h, h16, N * 32, 32);

  // Fused reconstruction + scorer heads.
  mlp_fused<<<gemm_bx, TB, 0, stream>>>(h16, prW1, rb1, prW2, rb2, psW1, sb1,
                                        sW2, sb2, out_xrec, out_scr, (int)N);
}
